// S2Site_18090402250770
// MI455X (gfx1250) — matrix operand order repaired and checked by reading, not compiled, not run on hardware
//
#include <hip/hip_runtime.h>
#include <hip/hip_bf16.h>

typedef __attribute__((ext_vector_type(16))) _Float16 v16h;
typedef __attribute__((ext_vector_type(8)))  float    v8f;

#define BB    8
#define NP    2048
#define NAA   512
#define KNN   16
#define K2    14
#define NG    64
#define DEMB  16
#define FDIM  128
#define EPSW  1e-6f

// ---------------- workspace layout (float offsets) ----------------
#define OFF_CENTER 0L          // 8*2048*3        = 49152
#define OFF_R      49152L      // 8*2048*9        = 147456
#define OFF_ATTR   196608L     // 8*2048*16       = 262144
#define OFF_ATT    458752L     // 8*2048          = 16384
#define OFF_FEAT   475136L     // 8*2048*128      = 2097152
#define OFF_POOLED 2572288L    // 8*512*128       = 524288
#define OFF_STATS  3096576L    // 256 (mean, rstd)
#define OFF_NBR    3096832L    // int  8*2048*16  = 262144
#define OFF_WFH    3358976L    // halves: 131072 + 16384 -> 73728 float slots

// Inverse of the ISA 16-bit A/B fragment maps (wave32).
// Given K-offset r within a 32-wide chunk: which half-lane group and which v16h slot.
__device__ __forceinline__ int frag_hi(int r) { return (r >> 3) & 1; }
__device__ __forceinline__ int frag_j(int r)  { return (r & 7) | ((r & 16) >> 1); }

// ---------------- kernel 0: convert + swizzle weights into fragment order ----------------
// GEMM1 B fragments: idx = (((nt*32 + kb)*32 + lane)*16 + j), value = W[kb*32 + ((j&8)<<1) + hi*8 + (j&7)][nt*16 + nl]
__global__ __launch_bounds__(256) void k0_convert(const float* __restrict__ Wf,
                                                  const float* __restrict__ Wfeat,
                                                  _Float16* __restrict__ WfhP,
                                                  _Float16* __restrict__ WfeathP) {
    long id = (long)blockIdx.x * 256 + threadIdx.x;
    if (id < NG * DEMB * FDIM) {              // 131072
        int j    = (int)(id & 15);
        int lane = (int)((id >> 4) & 31);
        int kb   = (int)((id >> 9) & 31);
        int nt   = (int)((id >> 14) & 7);
        int hi = lane >> 4, nl = lane & 15;
        int k = kb * 32 + ((j & 8) << 1) + hi * 8 + (j & 7);
        int n = nt * 16 + nl;
        WfhP[id] = (_Float16)Wf[(long)k * FDIM + n];
    }
    if (id < FDIM * FDIM) {                   // 16384
        int j    = (int)(id & 15);
        int lane = (int)((id >> 4) & 31);
        int kb   = (int)((id >> 9) & 3);
        int nt   = (int)((id >> 11) & 7);
        int hi = lane >> 4, nl = lane & 15;
        int k = kb * 32 + ((j & 8) << 1) + hi * 8 + (j & 7);
        int n = nt * 16 + nl;
        WfeathP[id] = (_Float16)Wfeat[(long)k * FDIM + n];
    }
}

// ---------------- kernel 1: local frames + embedded attrs ----------------
__global__ __launch_bounds__(256) void k1_frames(const int* __restrict__ fidx,
                                                 const int* __restrict__ attr_idx,
                                                 const float* __restrict__ pc,
                                                 const float* __restrict__ mframe,
                                                 const float* __restrict__ embed,
                                                 float* __restrict__ center,
                                                 float* __restrict__ Rm,
                                                 float* __restrict__ attr) {
    long gid = (long)blockIdx.x * 256 + threadIdx.x;
    if (gid >= (long)BB * NP) return;
    long bb = gid / NP;
    long pbase = bb * NP;
    int i0 = fidx[gid * 3 + 0], i1 = fidx[gid * 3 + 1], i2 = fidx[gid * 3 + 2];
    float cx = pc[(pbase + i0) * 3 + 0], cy = pc[(pbase + i0) * 3 + 1], cz = pc[(pbase + i0) * 3 + 2];
    float u1x = pc[(pbase + i1) * 3 + 0] - cx, u1y = pc[(pbase + i1) * 3 + 1] - cy, u1z = pc[(pbase + i1) * 3 + 2] - cz;
    float u2x = pc[(pbase + i2) * 3 + 0] - cx, u2y = pc[(pbase + i2) * 3 + 1] - cy, u2z = pc[(pbase + i2) * 3 + 2] - cz;
    float n1 = sqrtf(u1x * u1x + u1y * u1y + u1z * u1z) + EPSW;
    float e1x = u1x / n1, e1y = u1y / n1, e1z = u1z / n1;
    float nx = u1y * u2z - u1z * u2y, ny = u1z * u2x - u1x * u2z, nz = u1x * u2y - u1y * u2x;
    float nn = sqrtf(nx * nx + ny * ny + nz * nz) + EPSW;
    float e3x = nx / nn, e3y = ny / nn, e3z = nz / nn;
    float e2x = e3y * e1z - e3z * e1y, e2y = e3z * e1x - e3x * e1z, e2z = e3x * e1y - e3y * e1x;
    center[gid * 3 + 0] = cx; center[gid * 3 + 1] = cy; center[gid * 3 + 2] = cz;
    Rm[gid * 9 + 0] = e1x; Rm[gid * 9 + 1] = e1y; Rm[gid * 9 + 2] = e1z;
    Rm[gid * 9 + 3] = e2x; Rm[gid * 9 + 4] = e2y; Rm[gid * 9 + 5] = e2z;
    Rm[gid * 9 + 6] = e3x; Rm[gid * 9 + 7] = e3y; Rm[gid * 9 + 8] = e3z;
    int a = attr_idx[gid];
    float mf = mframe[gid];
#pragma unroll
    for (int d = 0; d < DEMB; ++d) attr[gid * DEMB + d] = embed[a * DEMB + d] * mf;
}

// ---------------- kernel 2: KNN (top-16 by squared distance) ----------------
__global__ __launch_bounds__(256) void k2_knn(const float* __restrict__ center,
                                              int* __restrict__ nbr) {
    __shared__ float cx[NP], cy[NP], cz[NP];
    int bb = blockIdx.x >> 3;
    int q0 = (blockIdx.x & 7) * 256;
    long base = (long)bb * NP;
    for (int j = threadIdx.x; j < NP; j += 256) {
        cx[j] = center[(base + j) * 3 + 0];
        cy[j] = center[(base + j) * 3 + 1];
        cz[j] = center[(base + j) * 3 + 2];
    }
    __syncthreads();
    int q = q0 + threadIdx.x;
    float qx = cx[q], qy = cy[q], qz = cz[q];
    float bd[KNN]; int bi[KNN];
#pragma unroll
    for (int k = 0; k < KNN; ++k) { bd[k] = 3.4e38f; bi[k] = 0; }
    for (int j = 0; j < NP; ++j) {
        float dx = cx[j] - qx, dy = cy[j] - qy, dz = cz[j] - qz;
        float d2 = dx * dx + dy * dy + dz * dz;
        if (d2 < bd[KNN - 1]) {
            int pos = KNN - 1;
#pragma unroll
            for (int s = KNN - 1; s > 0; --s) {
                if (d2 < bd[s - 1]) { bd[s] = bd[s - 1]; bi[s] = bi[s - 1]; pos = s - 1; }
            }
            bd[pos] = d2; bi[pos] = j;
        }
    }
#pragma unroll
    for (int k = 0; k < KNN; ++k) nbr[(base + q) * KNN + k] = bi[k];
}

// ---------------- kernel 3: fused gauss/Z + WMMA GEMMs (y, att, feat) ----------------
__global__ __launch_bounds__(128) void k3_fused(const float* __restrict__ center,
                                                const float* __restrict__ Rm,
                                                const float* __restrict__ attr,
                                                const int* __restrict__ nbr,
                                                const float* __restrict__ gmu,
                                                const float* __restrict__ gsig,
                                                const _Float16* __restrict__ WfhP,
                                                const _Float16* __restrict__ WfeathP,
                                                const float* __restrict__ Watt,
                                                const float* __restrict__ mframe,
                                                float* __restrict__ att_out,
                                                float* __restrict__ feat_out) {
    __shared__ float smem[13504];                         // 54016 B static LDS
    _Float16* ZhP = (_Float16*)smem;                      // A-fragment-packed Z: 32 chunks*32 lanes*16h = 32KB
    float* scratch = smem + 8192;
    // phase-A views
    float* localc = scratch;                              // 16*16*3
    float* attrN  = scratch + 768;                        // 16*16*16
    float* gp     = scratch + 768 + 4096;                 // 64*7
    // phase-B views (aliased after barrier)
    float*    yls = scratch;                              // 16*128 f32 row-major (for att)
    _Float16* yhP = (_Float16*)(scratch + 2048);          // A-fragment-packed y: 4*32*16h = 4KB

    const int tid = threadIdx.x;
    const int bb = blockIdx.x >> 7;
    const int tile = blockIdx.x & 127;
    const int i0 = tile * 16;
    const long base = (long)bb * NP;

    // gaussian params: inv_s2, mu*inv_s2, const term
    if (tid < NG) {
        float c = 0.f;
#pragma unroll
        for (int d = 0; d < 3; ++d) {
            float s = gsig[tid * 3 + d];
            float mu = gmu[tid * 3 + d];
            float iv = 1.0f / (s * s + EPSW);
            gp[tid * 7 + d] = iv;
            gp[tid * 7 + 3 + d] = mu * iv;
            c += mu * mu * iv;
        }
        gp[tid * 7 + 6] = c;
    }
    // phase A1: local coords + gathered attrs
    for (int pk = tid; pk < 256; pk += 128) {
        int p = pk >> 4, k = pk & 15;
        long gi = base + i0 + p;
        int j = nbr[gi * KNN + k];
        long gj = base + j;
        float dx = center[gj * 3 + 0] - center[gi * 3 + 0];
        float dy = center[gj * 3 + 1] - center[gi * 3 + 1];
        float dz = center[gj * 3 + 2] - center[gi * 3 + 2];
#pragma unroll
        for (int r = 0; r < 3; ++r)
            localc[pk * 3 + r] = dx * Rm[gi * 9 + r * 3 + 0] + dy * Rm[gi * 9 + r * 3 + 1] + dz * Rm[gi * 9 + r * 3 + 2];
#pragma unroll
        for (int d = 0; d < DEMB; ++d) attrN[pk * DEMB + d] = attr[gj * DEMB + d];
    }
    __syncthreads();
    // phase A2: Z[p][n*16+d] = sum_k gauss * attr  -> f16, scattered into A-fragment layout
    for (int rr = 0; rr < 8; ++rr) {
        int task = rr * 128 + tid;
        int p = task >> 6, n = task & 63;
        float is0 = gp[n * 7 + 0], is1 = gp[n * 7 + 1], is2 = gp[n * 7 + 2];
        float m0 = gp[n * 7 + 3], m1 = gp[n * 7 + 4], m2 = gp[n * 7 + 5];
        float cc = gp[n * 7 + 6];
        float acc[DEMB];
#pragma unroll
        for (int d = 0; d < DEMB; ++d) acc[d] = 0.f;
#pragma unroll
        for (int k = 0; k < KNN; ++k) {
            const float* l = &localc[(p * 16 + k) * 3];
            float q = l[0] * l[0] * is0 + l[1] * l[1] * is1 + l[2] * l[2] * is2
                    - 2.0f * (l[0] * m0 + l[1] * m1 + l[2] * m2) + cc;
            float g = __expf(-0.5f * q);
            const float* an = &attrN[(p * 16 + k) * DEMB];
#pragma unroll
            for (int d = 0; d < DEMB; ++d) acc[d] += g * an[d];
        }
        int kb = n >> 1;
        int rbase = (n & 1) << 4;
#pragma unroll
        for (int d = 0; d < DEMB; ++d) {
            int r = rbase + d;
            int laneT = (frag_hi(r) << 4) | p;
            ZhP[(((kb << 5) + laneT) << 4) + frag_j(r)] = (_Float16)acc[d];
        }
    }
    __syncthreads();

    const int lane = tid & 31, wv = tid >> 5;
    const int ml = lane & 15, hi = lane >> 4;
    const int nt0 = wv * 2, nt1 = wv * 2 + 1;
    const int ncol0 = nt0 * 16 + ml, ncol1 = nt1 * 16 + ml;
    const v16h* WfV    = (const v16h*)WfhP;
    const v16h* WfeatV = (const v16h*)WfeathP;

    // phase B: y = relu(Z @ W_filter) * mframe  (K=1024; A frag shared by both N tiles)
    {
        v8f c0 = {}, c1 = {};
        for (int kb = 0; kb < 32; ++kb) {
            v16h a  = *(const v16h*)&ZhP[((kb << 5) + lane) << 4];
            v16h b0 = WfV[((nt0 << 5) + kb) * 32 + lane];
            v16h b1 = WfV[((nt1 << 5) + kb) * 32 + lane];
            if (kb + 1 < 32) __builtin_prefetch(&WfV[((nt0 << 5) + kb + 1) * 32 + lane], 0, 0);
            c0 = __builtin_amdgcn_wmma_f32_16x16x32_f16(false, a, false, b0, (short)0, c0, false, false);
            c1 = __builtin_amdgcn_wmma_f32_16x16x32_f16(false, a, false, b1, (short)0, c1, false, false);
        }
#pragma unroll
        for (int j = 0; j < 8; ++j) {
            int mm = j + 8 * hi;
            float mf = mframe[base + i0 + mm];
            float v0 = fmaxf(c0[j], 0.f) * mf;
            float v1 = fmaxf(c1[j], 0.f) * mf;
            yls[mm * FDIM + ncol0] = v0;
            yls[mm * FDIM + ncol1] = v1;
            // scatter into A-fragment layout for GEMM2
            int r0 = ncol0 & 31, r1 = ncol1 & 31;
            yhP[((((ncol0 >> 5) << 5) + (frag_hi(r0) << 4) + mm) << 4) + frag_j(r0)] = (_Float16)v0;
            yhP[((((ncol1 >> 5) << 5) + (frag_hi(r1) << 4) + mm) << 4) + frag_j(r1)] = (_Float16)v1;
        }
    }
    __syncthreads();
    // phase C: att = (y @ W_att) * mframe
    if (tid < 16) {
        float s = 0.f;
#pragma unroll 4
        for (int f = 0; f < FDIM; ++f) s += yls[tid * FDIM + f] * Watt[f];
        att_out[base + i0 + tid] = s * mframe[base + i0 + tid];
    }
    __syncthreads();
    // phase D: feat = (y @ W_feat) * mframe  (K=128)
    {
        v8f c0 = {}, c1 = {};
#pragma unroll
        for (int kb = 0; kb < 4; ++kb) {
            v16h a  = *(const v16h*)&yhP[((kb << 5) + lane) << 4];
            v16h b0 = WfeatV[((nt0 << 2) + kb) * 32 + lane];
            v16h b1 = WfeatV[((nt1 << 2) + kb) * 32 + lane];
            c0 = __builtin_amdgcn_wmma_f32_16x16x32_f16(false, a, false, b0, (short)0, c0, false, false);
            c1 = __builtin_amdgcn_wmma_f32_16x16x32_f16(false, a, false, b1, (short)0, c1, false, false);
        }
#pragma unroll
        for (int j = 0; j < 8; ++j) {
            int mm = j + 8 * hi;
            float mf = mframe[base + i0 + mm];
            feat_out[(base + i0 + mm) * FDIM + ncol0] = c0[j] * mf;
            feat_out[(base + i0 + mm) * FDIM + ncol1] = c1[j] * mf;
        }
    }
}

// ---------------- kernel 4: per-residue top-14 + gated softmax pooling ----------------
__global__ __launch_bounds__(256) void k4_pool(const int* __restrict__ seq_atom,
                                               const int* __restrict__ seq_aa,
                                               const float* __restrict__ att,
                                               const float* __restrict__ feat,
                                               const float* __restrict__ mseq,
                                               float* __restrict__ pooled) {
    __shared__ int satom[NP];
    int bb = blockIdx.x >> 1;
    int a0 = (blockIdx.x & 1) * 256;
    for (int j = threadIdx.x; j < NP; j += 256) satom[j] = seq_atom[(long)bb * NP + j];
    __syncthreads();
    int a = a0 + threadIdx.x;
    int aval = seq_aa[(long)bb * NAA + a];
    float bd[K2]; int bi[K2];
#pragma unroll
    for (int k = 0; k < K2; ++k) { bd[k] = 3.4e38f; bi[k] = 0; }
    for (int j = 0; j < NP; ++j) {
        int di = aval - satom[j];
        float df = fabsf((float)di);
        if (df < bd[K2 - 1]) {
            int pos = K2 - 1;
#pragma unroll
            for (int s = K2 - 1; s > 0; --s) {
                if (df < bd[s - 1]) { bd[s] = bd[s - 1]; bi[s] = bi[s - 1]; pos = s - 1; }
            }
            bd[pos] = df; bi[pos] = j;
        }
    }
    float attv[K2], maxv = -3.4e38f;
#pragma unroll
    for (int k = 0; k < K2; ++k) {
        attv[k] = att[(long)bb * NP + bi[k]];
        maxv = fmaxf(maxv, attv[k]);
    }
    float w[K2], sum = 0.f;
#pragma unroll
    for (int k = 0; k < K2; ++k) {
        float gate = 1.0f - fminf(fmaxf(bd[k], 0.f), 1.f);
        w[k] = gate * __expf(attv[k] - maxv);
        sum += w[k];
    }
    float inv = 1.0f / (sum + EPSW);
#pragma unroll
    for (int k = 0; k < K2; ++k) w[k] *= inv;
    float m = mseq[(long)bb * NAA + a];
    long obase = ((long)bb * NAA + a) * FDIM;
    for (int fc = 0; fc < FDIM; fc += 4) {
        float s0 = 0.f, s1 = 0.f, s2 = 0.f, s3 = 0.f;
#pragma unroll
        for (int k = 0; k < K2; ++k) {
            const float* fp = feat + ((long)bb * NP + bi[k]) * FDIM + fc;
            s0 += w[k] * fp[0]; s1 += w[k] * fp[1]; s2 += w[k] * fp[2]; s3 += w[k] * fp[3];
        }
        pooled[obase + fc + 0] = s0 * m;
        pooled[obase + fc + 1] = s1 * m;
        pooled[obase + fc + 2] = s2 * m;
        pooled[obase + fc + 3] = s3 * m;
    }
}

// ---------------- kernel 5: masked mean/var per feature ----------------
__global__ __launch_bounds__(256) void k5_stats(const float* __restrict__ pooled,
                                                const float* __restrict__ mseq,
                                                float* __restrict__ stats) {
    __shared__ float s1s[256], s2s[256], sms[256];
    int f = blockIdx.x;
    float s1 = 0.f, s2 = 0.f, sm = 0.f;
    for (int row = threadIdx.x; row < BB * NAA; row += 256) {
        float v = pooled[(long)row * FDIM + f];
        float m = mseq[row];
        s1 += v * m; s2 += v * v * m; sm += m;
    }
    s1s[threadIdx.x] = s1; s2s[threadIdx.x] = s2; sms[threadIdx.x] = sm;
    __syncthreads();
    for (int off = 128; off > 0; off >>= 1) {
        if (threadIdx.x < off) {
            s1s[threadIdx.x] += s1s[threadIdx.x + off];
            s2s[threadIdx.x] += s2s[threadIdx.x + off];
            sms[threadIdx.x] += sms[threadIdx.x + off];
        }
        __syncthreads();
    }
    if (threadIdx.x == 0) {
        float msum = sms[0] + EPSW;
        float mean = s1s[0] / msum;
        float var = (s2s[0] - s1s[0] * s1s[0] / msum) / msum;
        stats[f] = mean;
        stats[FDIM + f] = rsqrtf(var + 1e-5f);
    }
}

// ---------------- kernel 6: batchnorm + relu + mask, emit outputs ----------------
__global__ __launch_bounds__(256) void k6_norm(const float* __restrict__ pooled,
                                               const float* __restrict__ mseq,
                                               const float* __restrict__ stats,
                                               const float* __restrict__ gamma,
                                               const float* __restrict__ beta,
                                               float* __restrict__ out) {
    long id = (long)blockIdx.x * 256 + threadIdx.x;   // < BB*NAA*FDIM
    int f = (int)(id & (FDIM - 1));
    long row = id >> 7;
    float v = (pooled[id] - stats[f]) * stats[FDIM + f] * gamma[f] + beta[f];
    float m = mseq[row];
    out[id] = fmaxf(v, 0.f) * m;
    if (f == 0) out[(long)BB * NAA * FDIM + row] = m;   // second tuple output: mseq_aa
}

extern "C" void kernel_launch(void* const* d_in, const int* in_sizes, int n_in,
                              void* d_out, int out_size, void* d_ws, size_t ws_size,
                              hipStream_t stream) {
    const int*   frame_indices = (const int*)d_in[0];
    const int*   attr_idx      = (const int*)d_in[1];
    const int*   seq_atom      = (const int*)d_in[2];
    const int*   seq_aa        = (const int*)d_in[3];
    const float* point_clouds  = (const float*)d_in[4];
    const float* mframe        = (const float*)d_in[5];
    const float* mseq_aa       = (const float*)d_in[6];
    const float* embed_table   = (const float*)d_in[7];
    const float* gauss_mu      = (const float*)d_in[8];
    const float* gauss_sigma   = (const float*)d_in[9];
    const float* W_filter      = (const float*)d_in[10];
    const float* W_att         = (const float*)d_in[11];
    const float* W_feat        = (const float*)d_in[12];
    const float* bn_gamma      = (const float*)d_in[13];
    const float* bn_beta       = (const float*)d_in[14];

    float* ws = (float*)d_ws;
    float* center = ws + OFF_CENTER;
    float* Rm     = ws + OFF_R;
    float* attr   = ws + OFF_ATTR;
    float* att    = ws + OFF_ATT;
    float* feat   = ws + OFF_FEAT;
    float* pooled = ws + OFF_POOLED;
    float* stats  = ws + OFF_STATS;
    int*   nbr    = (int*)(ws + OFF_NBR);
    _Float16* WfhP    = (_Float16*)(ws + OFF_WFH);
    _Float16* WfeathP = WfhP + NG * DEMB * FDIM;

    float* out = (float*)d_out;

    k0_convert<<<512, 256, 0, stream>>>(W_filter, W_feat, WfhP, WfeathP);
    k1_frames<<<64, 256, 0, stream>>>(frame_indices, attr_idx, point_clouds, mframe,
                                      embed_table, center, Rm, attr);
    k2_knn<<<64, 256, 0, stream>>>(center, nbr);
    k3_fused<<<BB * (NP / 16), 128, 0, stream>>>(center, Rm, attr, nbr, gauss_mu, gauss_sigma,
                                                 WfhP, WfeathP, W_att, mframe, att, feat);
    k4_pool<<<BB * (NAA / 256), 256, 0, stream>>>(seq_atom, seq_aa, att, feat, mseq_aa, pooled);
    k5_stats<<<FDIM, 256, 0, stream>>>(pooled, mseq_aa, stats);
    k6_norm<<<(BB * NAA * FDIM) / 256, 256, 0, stream>>>(pooled, mseq_aa, stats,
                                                         bn_gamma, bn_beta, out);
}
